// GCN_10222022164972
// MI455X (gfx1250) — matrix operand order repaired and checked by reading, not compiled, not run on hardware
//
#include <hip/hip_runtime.h>
#include <cstdint>

// GCN 2-layer forward for MI455X (gfx1250, wave32).
// degree -> dinv -> GEMM1 (TDM tile DMA -> bf16 LDS -> WMMA) -> self-loop init
// -> edge scatter -> GEMM2 (fused bias+relu bf16 staging -> WMMA) -> self-loop
// init -> edge scatter -> bias + log_softmax.

#define IN_F 512
#define HID  256
#define NCLS 40

typedef __attribute__((ext_vector_type(16))) __bf16 v16bf;
typedef __attribute__((ext_vector_type(8)))  __bf16 v8bf;
typedef __attribute__((ext_vector_type(8)))  float  v8f;
typedef __attribute__((ext_vector_type(4)))  unsigned int u32x4;
typedef __attribute__((ext_vector_type(8)))  int i32x8;
typedef __attribute__((ext_vector_type(4)))  int i32x4;

#if defined(__gfx1250__) && __has_builtin(__builtin_amdgcn_tensor_load_to_lds) && \
    __has_builtin(__builtin_amdgcn_s_wait_tensorcnt)
#define USE_TDM 1
#else
#define USE_TDM 0
#endif

// ---------------------------------------------------------------- degree / dinv
__global__ __launch_bounds__(256)
void fill_ones_kernel(float* __restrict__ deg, int n) {
    for (int i = blockIdx.x * blockDim.x + threadIdx.x; i < n;
         i += gridDim.x * blockDim.x)
        deg[i] = 1.0f;   // self-loop contributes 1 to every node's degree
}

__global__ __launch_bounds__(256)
void degree_kernel(const long long* __restrict__ dst, float* __restrict__ deg, int E) {
    for (int e = blockIdx.x * blockDim.x + threadIdx.x; e < E;
         e += gridDim.x * blockDim.x)
        unsafeAtomicAdd(&deg[(int)dst[e]], 1.0f);
}

__global__ __launch_bounds__(256)
void dinv_kernel(float* __restrict__ deg, int n) {
    for (int i = blockIdx.x * blockDim.x + threadIdx.x; i < n;
         i += gridDim.x * blockDim.x)
        deg[i] = rsqrtf(fmaxf(deg[i], 1.0f));   // in-place: deg -> deg^{-1/2}
}

// ---------------------------------------------------------------- TDM tile DMA
#if USE_TDM
// DMA a 16 x IN_F fp32 tile (row-major, stride IN_F) from global to LDS.
__device__ __forceinline__ void tdm_load_tile(const float* gptr, unsigned lds_addr) {
    unsigned long long ga = (unsigned long long)(uintptr_t)gptr;
    u32x4 g0;
    g0[0] = 1u;                                   // count = 1 valid descriptor
    g0[1] = lds_addr;                             // lds_addr [63:32]
    g0[2] = (unsigned)(ga & 0xFFFFFFFFu);         // global_addr [95:64]
    g0[3] = (unsigned)((ga >> 32) & 0x1FFFFFFu)   // global_addr [120:96]
          | (2u << 30);                           // type = 2 ("image")
    i32x8 g1;
    g1[0] = 0x00020000;                           // data_size = 2 (4 bytes)
    g1[1] = (int)(IN_F << 16);                    // tensor_dim0 = 512 (lo16)
    g1[2] = (int)(16u << 16);                     // tensor_dim0 hi=0 | tensor_dim1 = 16
    g1[3] = (int)(IN_F << 16);                    // tensor_dim1 hi=0 | tile_dim0 = 512
    g1[4] = 16;                                   // tile_dim1 = 16, tile_dim2 = 0
    g1[5] = IN_F;                                 // tensor_dim0_stride = 512
    g1[6] = 0;                                    // stride hi | dim1_stride lo
    g1[7] = 0;
    i32x4 g2 = {0, 0, 0, 0};
    i32x4 g3 = {0, 0, 0, 0};
#if __clang_major__ >= 23
    i32x8 g4 = {0, 0, 0, 0, 0, 0, 0, 0};
    __builtin_amdgcn_tensor_load_to_lds(g0, g1, g2, g3, g4, 0);
#else
    __builtin_amdgcn_tensor_load_to_lds(g0, g1, g2, g3, 0);
#endif
}
#endif

// ---------------------------------------------------------------- GEMM1: h = x @ W1
// block = 512 threads = 16 waves; wave w owns n-tile w (16 cols of 256).
// x tile staged (TDM DMA when available) then converted once to padded bf16 LDS;
// A-fragments are built with two ds_load_b128, zero VALU in the WMMA loop.
__global__ __launch_bounds__(512)
void gemm1_kernel(const float* __restrict__ x, const float* __restrict__ W1,
                  float* __restrict__ h, int n_nodes) {
    constexpr int LDSB = IN_F + 8;              // 520 bf16: rows on distinct banks
    __shared__ __align__(16) __bf16 Sbf[16 * LDSB];
#if USE_TDM
    __shared__ __align__(16) float Sf32[16 * IN_F];
#endif

    const int tid   = threadIdx.x;
    const int wave  = tid >> 5;
    const int lane  = tid & 31;
    const int hl    = lane >> 4;                // half-wave (0/1)
    const int l16   = lane & 15;
    const int ntile = wave;                     // 0..15
    const int ncol  = ntile * 16 + l16;

    // Preload B fragments: element i of k-step ks is W1[K][ncol], K = ks*32+hl*8+i (i<8), ks*32+16+hl*8+(i-8) (i>=8)
    v16bf Bfrag[IN_F / 32];
#pragma unroll
    for (int ks = 0; ks < IN_F / 32; ++ks) {
        v16bf b;
#pragma unroll
        for (int i = 0; i < 16; ++i) {
            int k = ks * 32 + (i < 8 ? hl * 8 + i : 16 + hl * 8 + (i - 8));
            b[i] = (__bf16)W1[k * HID + ncol];
        }
        Bfrag[ks] = b;
    }

    const int mtiles = n_nodes / 16;            // 6250 exactly
    for (int mt = blockIdx.x; mt < mtiles; mt += gridDim.x) {
        const float* xt = x + (size_t)mt * 16 * IN_F;
#if USE_TDM
        if (wave == 0) {                        // one DMA per tile (EXEC ignored)
            tdm_load_tile(xt, (unsigned)(uintptr_t)(void*)Sf32);
            __builtin_amdgcn_s_wait_tensorcnt(0);
        }
        __syncthreads();                        // fp32 tile visible to all waves
        for (int idx = tid; idx < 16 * IN_F; idx += 512) {
            int r = idx >> 9;
            int c = idx & (IN_F - 1);
            Sbf[r * LDSB + c] = (__bf16)Sf32[r * IN_F + c];
        }
#else
        for (int idx = tid; idx < 16 * IN_F; idx += 512) {
            int r = idx >> 9;
            int c = idx & (IN_F - 1);
            Sbf[r * LDSB + c] = (__bf16)xt[r * IN_F + c];
        }
#endif
        int mt2 = mt + gridDim.x;               // prefetch next tile
        if (mt2 < mtiles)
            __builtin_prefetch(x + (size_t)mt2 * 16 * IN_F + tid * 16, 0, 3);
        __syncthreads();                        // bf16 tile ready

        v8f acc = {};
#pragma unroll
        for (int ks = 0; ks < IN_F / 32; ++ks) {
            // A elem i: K = ks*32 + hl*8 + i (i<8) ; K = ks*32 + 16 + hl*8 + (i-8)
            const __bf16* ar = &Sbf[l16 * LDSB + ks * 32 + hl * 8];
            v8bf lo = *(const v8bf*)ar;         // one ds_load_b128
            v8bf hi = *(const v8bf*)(ar + 16);  // one ds_load_b128
            v16bf a = __builtin_shufflevector(lo, hi, 0, 1, 2, 3, 4, 5, 6, 7,
                                              8, 9, 10, 11, 12, 13, 14, 15);
            acc = __builtin_amdgcn_wmma_f32_16x16x32_bf16(
                false, a, false, Bfrag[ks], (short)0, acc, false, false);
        }
        // D layout: VGPR r -> M = r + 8*hl, N = l16
        float* hp = h + ((size_t)mt * 16 + hl * 8) * HID + ntile * 16 + l16;
#pragma unroll
        for (int r = 0; r < 8; ++r) hp[(size_t)r * HID] = acc[r];
        __syncthreads();                        // all reads done before next tile
    }
}

// ---------------------------------------------------------------- GEMM2: g = relu(agg+b1) @ W2
// block = 96 threads = 3 waves; bias+relu fused into bf16 LDS staging.
__global__ __launch_bounds__(96)
void gemm2_kernel(const float* __restrict__ agg, const float* __restrict__ b1,
                  const float* __restrict__ W2, float* __restrict__ g, int n_nodes) {
    constexpr int K    = HID;                   // 256
    constexpr int LDSB = K + 8;                 // 264 bf16 row stride
    __shared__ __align__(16) __bf16 As[16 * LDSB];

    const int tid   = threadIdx.x;
    const int wave  = tid >> 5;
    const int lane  = tid & 31;
    const int hl    = lane >> 4;
    const int l16   = lane & 15;
    const int ntile = wave;                     // 0..2
    const int ncol  = ntile * 16 + l16;         // 0..47 (valid < 40)

    v16bf Bfrag[K / 32];
#pragma unroll
    for (int ks = 0; ks < K / 32; ++ks) {
        v16bf b;
#pragma unroll
        for (int i = 0; i < 16; ++i) {
            int k = ks * 32 + (i < 8 ? hl * 8 + i : 16 + hl * 8 + (i - 8));
            b[i] = (ncol < NCLS) ? (__bf16)W2[k * NCLS + ncol] : (__bf16)0.0f;
        }
        Bfrag[ks] = b;
    }

    const int mtiles = n_nodes / 16;
    for (int mt = blockIdx.x; mt < mtiles; mt += gridDim.x) {
        const float* at = agg + (size_t)mt * 16 * K;
        for (int idx = tid; idx < 16 * K; idx += 96) {
            int r = idx >> 8;
            int c = idx & (K - 1);
            As[r * LDSB + c] = (__bf16)fmaxf(at[r * K + c] + b1[c], 0.0f);
        }
        __syncthreads();

        v8f acc = {};
#pragma unroll
        for (int ks = 0; ks < K / 32; ++ks) {
            const __bf16* ar = &As[l16 * LDSB + ks * 32 + hl * 8];
            v8bf lo = *(const v8bf*)ar;
            v8bf hi = *(const v8bf*)(ar + 16);
            v16bf a = __builtin_shufflevector(lo, hi, 0, 1, 2, 3, 4, 5, 6, 7,
                                              8, 9, 10, 11, 12, 13, 14, 15);
            acc = __builtin_amdgcn_wmma_f32_16x16x32_bf16(
                false, a, false, Bfrag[ks], (short)0, acc, false, false);
        }
        if (ncol < NCLS) {
            float* gp = g + ((size_t)mt * 16 + hl * 8) * NCLS + ncol;
#pragma unroll
            for (int r = 0; r < 8; ++r) gp[(size_t)r * NCLS] = acc[r];
        }
        __syncthreads();
    }
}

// ---------------------------------------------------------------- self-loop init: agg = dinv^2 * h
__global__ __launch_bounds__(256)
void selfloop_init_kernel(const float* __restrict__ hin, const float* __restrict__ dinv,
                          float* __restrict__ agg, int n, int F) {
    size_t total = (size_t)n * F;
    for (size_t i = blockIdx.x * (size_t)blockDim.x + threadIdx.x; i < total;
         i += (size_t)gridDim.x * blockDim.x) {
        int node = (int)(i / F);
        float dv = dinv[node];
        agg[i] = hin[i] * dv * dv;
    }
}

// ---------------------------------------------------------------- edge scatter (one wave per edge)
template <int F>
__global__ __launch_bounds__(256)
void propagate_kernel(const long long* __restrict__ src, const long long* __restrict__ dst,
                      const float* __restrict__ dinv, const float* __restrict__ hin,
                      float* __restrict__ agg, int E) {
    const int wid    = (int)((blockIdx.x * (size_t)blockDim.x + threadIdx.x) >> 5);
    const int lane   = threadIdx.x & 31;
    const int nwaves = (gridDim.x * blockDim.x) >> 5;
    for (int e = wid; e < E; e += nwaves) {
        int s = (int)src[e];
        int d = (int)dst[e];
        float norm = dinv[s] * dinv[d];
        const float* hs = hin + (size_t)s * F;
        float*       ad = agg + (size_t)d * F;
#pragma unroll
        for (int c = lane; c < F; c += 32)
            unsafeAtomicAdd(&ad[c], hs[c] * norm);
    }
}

// ---------------------------------------------------------------- bias + log_softmax (one wave/node)
__global__ __launch_bounds__(256)
void logsoftmax_kernel(const float* __restrict__ agg2, const float* __restrict__ b2,
                       float* __restrict__ out, int n) {
    const int wid  = (int)((blockIdx.x * (size_t)blockDim.x + threadIdx.x) >> 5);
    const int lane = threadIdx.x & 31;
    if (wid >= n) return;
    const float* z = agg2 + (size_t)wid * NCLS;

    float v0 = z[lane] + b2[lane];                                  // lanes 0..31
    float v1 = (lane + 32 < NCLS) ? (z[lane + 32] + b2[lane + 32])  // lanes 0..7
                                  : -__builtin_inff();
    float m = fmaxf(v0, v1);
#pragma unroll
    for (int off = 16; off >= 1; off >>= 1) m = fmaxf(m, __shfl_xor(m, off));

    float s = __expf(v0 - m) + ((lane + 32 < NCLS) ? __expf(v1 - m) : 0.0f);
#pragma unroll
    for (int off = 16; off >= 1; off >>= 1) s += __shfl_xor(s, off);

    float lse = m + __logf(s);
    float* o = out + (size_t)wid * NCLS;
    o[lane] = v0 - lse;
    if (lane + 32 < NCLS) o[lane + 32] = v1 - lse;
}

// ---------------------------------------------------------------- launch
extern "C" void kernel_launch(void* const* d_in, const int* in_sizes, int n_in,
                              void* d_out, int out_size, void* d_ws, size_t ws_size,
                              hipStream_t stream) {
    const float*     x   = (const float*)d_in[0];
    const long long* ei  = (const long long*)d_in[1];   // int64 per reference
    const float*     W1  = (const float*)d_in[2];
    const float*     b1  = (const float*)d_in[3];
    const float*     W2  = (const float*)d_in[4];
    const float*     b2  = (const float*)d_in[5];
    float*           out = (float*)d_out;

    const int n = in_sizes[0] / IN_F;       // 100000
    const int E = in_sizes[1] / 2;          // 3200000
    const long long* src = ei;
    const long long* dst = ei + E;

    // workspace layout (floats): [dinv n][h n*HID (reused: g, agg2)][agg n*HID]
    float* ws   = (float*)d_ws;
    float* dinv = ws;
    float* h    = ws + n;
    float* agg  = h + (size_t)n * HID;
    float* g    = h;                        // h is dead after propagate-1
    float* agg2 = h + (size_t)n * NCLS;     // disjoint from g, inside h region

    const int t256_n = (n + 255) / 256;

    // degree (with self-loops) -> dinv, in place
    fill_ones_kernel<<<t256_n, 256, 0, stream>>>(dinv, n);
    degree_kernel<<<2048, 256, 0, stream>>>(dst, dinv, E);
    dinv_kernel<<<t256_n, 256, 0, stream>>>(dinv, n);

    // layer 1
    gemm1_kernel<<<1024, 512, 0, stream>>>(x, W1, h, n);
    selfloop_init_kernel<<<4096, 256, 0, stream>>>(h, dinv, agg, n, HID);
    propagate_kernel<HID><<<4096, 256, 0, stream>>>(src, dst, dinv, h, agg, E);

    // layer 2 (bias+relu fused into GEMM2 staging)
    gemm2_kernel<<<2048, 96, 0, stream>>>(agg, b1, W2, g, n);
    selfloop_init_kernel<<<2048, 256, 0, stream>>>(g, dinv, agg2, n, NCLS);
    propagate_kernel<NCLS><<<4096, 256, 0, stream>>>(src, dst, dinv, g, agg2, E);

    // bias + log_softmax -> d_out
    logsoftmax_kernel<<<(n * 32 + 255) / 256, 256, 0, stream>>>(agg2, b2, out, n);
}
